// TruePsiQRHTransformer_34299608826344
// MI455X (gfx1250) — matrix operand order repaired and checked by reading, not compiled, not run on hardware
//
#include <hip/hip_runtime.h>
#include <hip/hip_bf16.h>
#include <math.h>

#define L_LAYERS 4
#define D_MODEL  1024
#define QD_      (D_MODEL / 4)
#define B_       8
#define T_       2048
#define NUM_CLS  1000
#define EPS_LN   1e-5f
#define EPS_FILT 1e-8f
#define BT       (B_ * T_)

typedef __bf16 bf16_t;
typedef __attribute__((ext_vector_type(16))) __bf16 v16bf;
typedef __attribute__((ext_vector_type(8)))  __bf16 v8bf;
typedef __attribute__((ext_vector_type(8)))  float  v8f;

union V16U { v16bf v; v8bf h[2]; };

// ---------------------------------------------------------------------------
// CDNA5 async global->LDS DMA (ASYNCcnt-tracked), 16B per lane.
// LDS operand: flat addresses in the LDS aperture carry the LDS byte offset
// in bits[31:0] (ISA 10.2), so truncating the generic pointer is the address.
// ---------------------------------------------------------------------------
__device__ __forceinline__ void async_copy16(void* lds_ptr, const void* gptr) {
  unsigned lds_addr = (unsigned)(size_t)lds_ptr;
  unsigned long long gaddr = (unsigned long long)(size_t)gptr;
  asm volatile("global_load_async_to_lds_b128 %0, %1, off"
               :: "v"(lds_addr), "v"(gaddr) : "memory");
}
// ASYNCcnt decrements in issue order for async loads, so waiting <=4 with two
// 4-transfer tiles outstanding guarantees the OLDEST tile has landed.
__device__ __forceinline__ void wait_async_le(int tiles_in_flight) {
  if (tiles_in_flight > 0)
    asm volatile("s_wait_asynccnt 0x4" ::: "memory");
  else
    asm volatile("s_wait_asynccnt 0x0" ::: "memory");
}

// ---------------------------------------------------------------------------
// f32 -> bf16 conversion (weights once per launch; activations fused elsewhere)
// ---------------------------------------------------------------------------
__global__ void f2bf_kernel(const float* __restrict__ src, bf16_t* __restrict__ dst, long n) {
  long i = blockIdx.x * (long)blockDim.x + threadIdx.x;
  long st = (long)gridDim.x * blockDim.x;
  for (; i < n; i += st) dst[i] = (bf16_t)src[i];
}

// ---------------------------------------------------------------------------
// Embedding gather + positional add; emits f32 x and bf16 copy for WMMA A.
// ---------------------------------------------------------------------------
__global__ void embed_kernel(const int* __restrict__ tok, const float* __restrict__ emb,
                             const float* __restrict__ pos, float* __restrict__ x,
                             bf16_t* __restrict__ xbf, long n) {
  long i = blockIdx.x * (long)blockDim.x + threadIdx.x;
  long st = (long)gridDim.x * blockDim.x;
  for (; i < n; i += st) {
    long bt = i / D_MODEL;
    int d = (int)(i - bt * D_MODEL);
    int t = (int)(bt % T_);
    float v = emb[(size_t)tok[bt] * D_MODEL + d] + pos[(size_t)t * D_MODEL + d];
    x[i] = v;
    xbf[i] = (bf16_t)v;
  }
}

// ---------------------------------------------------------------------------
// Fused Q/R/H bf16 WMMA GEMM:  C[m,n] = sum_k A[m,k]*W[n,k] + bias[n]
// gridDim.z in {0,1,2} selects Wq/Wr/Wh. Block 128x128, K-step 32, 8 waves.
// Triple-buffered LDS filled by GLOBAL_LOAD_ASYNC_TO_LDS_B128 with TWO tiles
// of DMA in flight behind the 8 WMMAs of the current tile. Row pitch 40 bf16
// (80B): 16B alignment, 20-bank row advance, conflict-free fragment ds_loads.
// ---------------------------------------------------------------------------
#define TM 128
#define TN 128
#define TK 32
#define LDSR 40

__global__ __launch_bounds__(256) void gemm_qrh_wmma(
    const bf16_t* __restrict__ A,
    const bf16_t* __restrict__ W0, const bf16_t* __restrict__ W1, const bf16_t* __restrict__ W2,
    const float* __restrict__ b0, const float* __restrict__ b1, const float* __restrict__ b2,
    float* __restrict__ C0, float* __restrict__ C1, float* __restrict__ C2,
    int M, int N, int K)
{
  __shared__ __align__(16) bf16_t As[3][TM * LDSR];
  __shared__ __align__(16) bf16_t Bs[3][TN * LDSR];

  const bf16_t* Bw; const float* bias; float* C;
  if (blockIdx.z == 0)      { Bw = W0; bias = b0; C = C0; }
  else if (blockIdx.z == 1) { Bw = W1; bias = b1; C = C1; }
  else                      { Bw = W2; bias = b2; C = C2; }

  const int tid  = threadIdx.x;
  const int wave = tid >> 5, lane = tid & 31;
  const int lr   = lane & 15, half = lane >> 4;
  const int waveM = wave & 1, waveN = wave >> 1;   // 2 x 4 wave grid
  const int m0 = blockIdx.x * TM, n0 = blockIdx.y * TN;

  // Tile fill: 128 rows x 32 bf16 = 512 16B chunks per matrix; 2 per thread.
  const int r0 = tid >> 2,         c0 = (tid & 3) << 3;
  const int r1 = (tid + 256) >> 2, c1 = ((tid + 256) & 3) << 3;

  auto issue = [&](int buf, int k0) {
    async_copy16(&As[buf][r0 * LDSR + c0], A  + (size_t)(m0 + r0) * K + k0 + c0);
    async_copy16(&As[buf][r1 * LDSR + c1], A  + (size_t)(m0 + r1) * K + k0 + c1);
    async_copy16(&Bs[buf][r0 * LDSR + c0], Bw + (size_t)(n0 + r0) * K + k0 + c0);
    async_copy16(&Bs[buf][r1 * LDSR + c1], Bw + (size_t)(n0 + r1) * K + k0 + c1);
  };

  const v8f vzero = {0.f, 0.f, 0.f, 0.f, 0.f, 0.f, 0.f, 0.f};
  v8f acc[4][2];
#pragma unroll
  for (int mf = 0; mf < 4; ++mf)
#pragma unroll
    for (int nf = 0; nf < 2; ++nf) acc[mf][nf] = vzero;

  const int nk = K / TK;
  issue(0, 0);
  if (nk > 1) issue(1, TK);

  int buf = 0;
  for (int t = 0; t < nk; ++t) {
    wait_async_le((t + 1 < nk) ? 1 : 0);  // oldest tile's DMA has landed in LDS
    __syncthreads();                      // all waves' DMA landed; old reads done
    if (t + 2 < nk) {
      int bn = buf + 2; if (bn >= 3) bn -= 3;
      issue(bn, (t + 2) * TK);            // keep two tiles of DMA in flight
    }

    // B frag: lane half 0 -> K 0..15, half 1 -> K 16..31 (contiguous 32B)
    V16U bfrag[2];
#pragma unroll
    for (int nf = 0; nf < 2; ++nf) {
      const bf16_t* bp = &Bs[buf][(waveN * 32 + nf * 16 + lr) * LDSR + half * 16];
      bfrag[nf].h[0] = *(const v8bf*)(bp);
      bfrag[nf].h[1] = *(const v8bf*)(bp + 8);
    }
    // A frag: lane half 0 -> K 0..15, half 1 -> K 16..31 (contiguous 32B)
    V16U afrag[4];
#pragma unroll
    for (int mf = 0; mf < 4; ++mf) {
      const bf16_t* ap = &As[buf][(waveM * 64 + mf * 16 + lr) * LDSR + half * 16];
      afrag[mf].h[0] = *(const v8bf*)(ap);
      afrag[mf].h[1] = *(const v8bf*)(ap + 8);
    }

#pragma unroll
    for (int mf = 0; mf < 4; ++mf)
#pragma unroll
      for (int nf = 0; nf < 2; ++nf)
        acc[mf][nf] = __builtin_amdgcn_wmma_f32_16x16x32_bf16(
            false, afrag[mf].v, false, bfrag[nf].v,
            (short)0, acc[mf][nf], false, false);

    buf = (buf == 2) ? 0 : buf + 1;
  }

  // Epilogue: VGPR r -> (M = r + 8*half_hi, N = lane%16)
#pragma unroll
  for (int nf = 0; nf < 2; ++nf) {
    const int n = n0 + waveN * 32 + nf * 16 + lr;
    const float bv = bias[n];
#pragma unroll
    for (int mf = 0; mf < 4; ++mf) {
      const int mb = m0 + waveM * 64 + mf * 16 + half * 8;
#pragma unroll
      for (int r = 0; r < 8; ++r)
        C[(size_t)(mb + r) * N + n] = acc[mf][nf][r] + bv;
    }
  }
}

// ---------------------------------------------------------------------------
// In-LDS radix-2 FFT, 2048 points, 256 threads, input in bit-reversed order.
// sgn = +1 forward (e^{-2pi i}), -1 inverse (unscaled).
// ---------------------------------------------------------------------------
#define FN 2048
#define FLOG 11

__device__ inline void fft2048(float2* s, int tid, float sgn) {
  for (int st = 1; st <= FLOG; ++st) {
    const int h = 1 << (st - 1);
    const int len = 1 << st;
    const float base = sgn * (-6.283185307179586f) / (float)len;
    for (int i = tid; i < FN / 2; i += 256) {
      const int j = i >> (st - 1);
      const int k = i & (h - 1);
      const int idx = j * len + k;
      float sn, cs;
      __sincosf(base * (float)k, &sn, &cs);
      float2 u = s[idx], v = s[idx + h];
      float2 tw = make_float2(v.x * cs - v.y * sn, v.x * sn + v.y * cs);
      s[idx]     = make_float2(u.x + tw.x, u.y + tw.y);
      s[idx + h] = make_float2(u.x - tw.x, u.y - tw.y);
    }
    __syncthreads();
  }
}

// ---------------------------------------------------------------------------
// Spectral cross-correlation: one (b, channel) column per block.
// Packed real trick: z = Q + iR -> one fwd FFT; unpack Qs,Rs; apply phase
// filter; S = Qf*conj(Rf); one inverse FFT; write Re to Q in place.
// ---------------------------------------------------------------------------
__global__ __launch_bounds__(256) void spectral_kernel(
    float* __restrict__ Q, const float* __restrict__ R,
    const float* __restrict__ alpha)
{
  __shared__ float2 Z[FN];
  __shared__ float2 S[FN];
  const int tid = threadIdx.x;
  const int b = blockIdx.x / D_MODEL;
  const int c = blockIdx.x - b * D_MODEL;
  const float a = alpha[c & (QD_ - 1)];
  const size_t base = (size_t)b * T_ * D_MODEL + c;

  for (int t = tid; t < FN; t += 256) {
    int rv = __brev((unsigned)t) >> (32 - FLOG);
    Z[rv] = make_float2(Q[base + (size_t)t * D_MODEL],
                        R[base + (size_t)t * D_MODEL]);
  }
  __syncthreads();
  fft2048(Z, tid, 1.0f);

  for (int k = tid; k < FN; k += 256) {
    int km = (FN - k) & (FN - 1);
    float2 zk = Z[k], zm = Z[km];
    float qx = 0.5f * (zk.x + zm.x), qy = 0.5f * (zk.y - zm.y);
    float wx = zk.x - zm.x,          wy = zk.y + zm.y;
    float rx = 0.5f * wy,            ry = -0.5f * wx;
    float pq = a * atanf(logf(sqrtf(qx * qx + qy * qy) + EPS_FILT));
    float pr = a * atanf(logf(sqrtf(rx * rx + ry * ry) + EPS_FILT));
    float cq, sq, cr, sr;
    __sincosf(pq, &sq, &cq);
    __sincosf(pr, &sr, &cr);
    float qfx = qx * cq - qy * sq, qfy = qx * sq + qy * cq;
    float rfx = rx * cr - ry * sr, rfy = rx * sr + ry * cr;
    float sx = qfx * rfx + qfy * rfy;       // Qf * conj(Rf)
    float sy = qfy * rfx - qfx * rfy;
    int rv = __brev((unsigned)k) >> (32 - FLOG);
    S[rv] = make_float2(sx, sy);
  }
  __syncthreads();
  fft2048(S, tid, -1.0f);

  const float invn = 1.0f / (float)FN;
  for (int t = tid; t < FN; t += 256)
    Q[base + (size_t)t * D_MODEL] = S[t].x * invn;
}

// ---------------------------------------------------------------------------
// Hamilton product (quaternion) + layernorm. Block per (b,t); thread = qd.
// ---------------------------------------------------------------------------
__global__ __launch_bounds__(256) void hamln_kernel(
    const float* __restrict__ inter, const float* __restrict__ H,
    const float* __restrict__ lnw, const float* __restrict__ lnb,
    float* __restrict__ y)
{
  __shared__ float2 red[256];
  const int tid = threadIdx.x;
  const size_t base = (size_t)blockIdx.x * D_MODEL;

  float w1 = inter[base + 0 * QD_ + tid], x1 = inter[base + 1 * QD_ + tid];
  float y1 = inter[base + 2 * QD_ + tid], z1 = inter[base + 3 * QD_ + tid];
  float w2 = H[base + 0 * QD_ + tid], x2 = H[base + 1 * QD_ + tid];
  float y2 = H[base + 2 * QD_ + tid], z2 = H[base + 3 * QD_ + tid];

  float o[4];
  o[0] = w1 * w2 - x1 * x2 - y1 * y2 - z1 * z2;
  o[1] = w1 * x2 + x1 * w2 + y1 * z2 - z1 * y2;
  o[2] = w1 * y2 - x1 * z2 + y1 * w2 + z1 * x2;
  o[3] = w1 * z2 + x1 * y2 - y1 * x2 + z1 * w2;

  float s = o[0] + o[1] + o[2] + o[3];
  float ss = o[0] * o[0] + o[1] * o[1] + o[2] * o[2] + o[3] * o[3];
  red[tid] = make_float2(s, ss);
  __syncthreads();
  for (int st = 128; st > 0; st >>= 1) {
    if (tid < st) { red[tid].x += red[tid + st].x; red[tid].y += red[tid + st].y; }
    __syncthreads();
  }
  float mean = red[0].x * (1.0f / D_MODEL);
  float var  = red[0].y * (1.0f / D_MODEL) - mean * mean;
  float rstd = rsqrtf(var + EPS_LN);
#pragma unroll
  for (int k = 0; k < 4; ++k) {
    int ci = k * QD_ + tid;
    y[base + ci] = (o[k] - mean) * rstd * lnw[ci] + lnb[ci];
  }
}

// ---------------------------------------------------------------------------
// Second spectral pass collapsed: ifft(fft(x)*gate) == gate*x (gate is
// frequency-independent per channel). Fused gate * rot-mix + residual + LN.
// Emits f32 x and bf16 copy for next layer's GEMMs.
// ---------------------------------------------------------------------------
__global__ __launch_bounds__(256) void grln_kernel(
    const float* __restrict__ y, const float* __restrict__ rot,
    const float* __restrict__ gate,
    const float* __restrict__ lnw, const float* __restrict__ lnb,
    float* __restrict__ x, bf16_t* __restrict__ xbf)
{
  __shared__ float2 red[256];
  const int tid = threadIdx.x;
  const size_t base = (size_t)blockIdx.x * D_MODEL;
  const float g = gate[tid];

  float xf[4];
#pragma unroll
  for (int k = 0; k < 4; ++k) xf[k] = y[base + k * QD_ + tid] * g;

  float v[4];
  float s = 0.f, ss = 0.f;
#pragma unroll
  for (int j = 0; j < 4; ++j) {
    float a = 0.f;
#pragma unroll
    for (int k = 0; k < 4; ++k) a += rot[j * 4 + k] * xf[k];
    a += x[base + j * QD_ + tid];      // residual
    v[j] = a; s += a; ss += a * a;
  }
  red[tid] = make_float2(s, ss);
  __syncthreads();
  for (int st = 128; st > 0; st >>= 1) {
    if (tid < st) { red[tid].x += red[tid + st].x; red[tid].y += red[tid + st].y; }
    __syncthreads();
  }
  float mean = red[0].x * (1.0f / D_MODEL);
  float var  = red[0].y * (1.0f / D_MODEL) - mean * mean;
  float rstd = rsqrtf(var + EPS_LN);
#pragma unroll
  for (int j = 0; j < 4; ++j) {
    int ci = j * QD_ + tid;
    float o = (v[j] - mean) * rstd * lnw[ci] + lnb[ci];
    x[base + ci]  = o;
    xbf[base + ci] = (bf16_t)o;
  }
}

// ---------------------------------------------------------------------------
// Mean pool over T.
// ---------------------------------------------------------------------------
__global__ __launch_bounds__(256) void pool_kernel(const float* __restrict__ x,
                                                   float* __restrict__ pooled) {
  const int b = blockIdx.x >> 2;
  const int d = ((blockIdx.x & 3) << 8) + threadIdx.x;
  const float* p = x + (size_t)b * T_ * D_MODEL + d;
  float s = 0.f;
  for (int t = 0; t < T_; ++t) s += p[(size_t)t * D_MODEL];
  pooled[b * D_MODEL + d] = s * (1.0f / T_);
}

// ---------------------------------------------------------------------------
// Final LN + classifier. Block per batch row; wave-parallel dot products.
// ---------------------------------------------------------------------------
__global__ __launch_bounds__(256) void cls_kernel(
    const float* __restrict__ pooled,
    const float* __restrict__ lnw, const float* __restrict__ lnb,
    const float* __restrict__ W, const float* __restrict__ bias,
    float* __restrict__ out)
{
  __shared__ float hbuf[D_MODEL];
  __shared__ float2 red[256];
  const int tid = threadIdx.x;
  const int b = blockIdx.x;

  float loc[4];
  float s = 0.f, ss = 0.f;
#pragma unroll
  for (int j = 0; j < 4; ++j) {
    float v = pooled[b * D_MODEL + j * 256 + tid];
    loc[j] = v; s += v; ss += v * v;
  }
  red[tid] = make_float2(s, ss);
  __syncthreads();
  for (int st = 128; st > 0; st >>= 1) {
    if (tid < st) { red[tid].x += red[tid + st].x; red[tid].y += red[tid + st].y; }
    __syncthreads();
  }
  float mean = red[0].x * (1.0f / D_MODEL);
  float var  = red[0].y * (1.0f / D_MODEL) - mean * mean;
  float rstd = rsqrtf(var + EPS_LN);
#pragma unroll
  for (int j = 0; j < 4; ++j) {
    int ci = j * 256 + tid;
    hbuf[ci] = (loc[j] - mean) * rstd * lnw[ci] + lnb[ci];
  }
  __syncthreads();

  const int wave = tid >> 5, lane = tid & 31;
  for (int n = wave; n < NUM_CLS; n += 8) {
    const float* wr = W + (size_t)n * D_MODEL;
    float acc = 0.f;
    for (int k = lane; k < D_MODEL; k += 32) acc += hbuf[k] * wr[k];
#pragma unroll
    for (int m = 16; m > 0; m >>= 1) acc += __shfl_xor(acc, m, 32);
    if (lane == 0) out[b * NUM_CLS + n] = acc + bias[n];
  }
}

// ---------------------------------------------------------------------------
extern "C" void kernel_launch(void* const* d_in, const int* in_sizes, int n_in,
                              void* d_out, int out_size, void* d_ws, size_t ws_size,
                              hipStream_t stream)
{
  const int*   tokens = (const int*)  d_in[0];
  const float* embw   = (const float*)d_in[1];
  const float* pos    = (const float*)d_in[2];
  const float* Wq     = (const float*)d_in[3];
  const float* bq     = (const float*)d_in[4];
  const float* Wr     = (const float*)d_in[5];
  const float* br     = (const float*)d_in[6];
  const float* Wh     = (const float*)d_in[7];
  const float* bh     = (const float*)d_in[8];
  const float* alpha  = (const float*)d_in[9];
  const float* spw    = (const float*)d_in[10];
  const float* spb    = (const float*)d_in[11];
  const float* gate   = (const float*)d_in[12];
  const float* rot    = (const float*)d_in[13];
  const float* olw    = (const float*)d_in[14];
  const float* olb    = (const float*)d_in[15];
  const float* clw    = (const float*)d_in[16];
  const float* clb    = (const float*)d_in[17];
  const float* clsW   = (const float*)d_in[18];
  const float* clsb   = (const float*)d_in[19];
  float* out = (float*)d_out;
  (void)in_sizes; (void)n_in; (void)out_size; (void)ws_size;

  char* ws = (char*)d_ws;
  size_t off = 0;
  auto alloc = [&](size_t bytes) -> void* {
    void* p = ws + off;
    off = (off + bytes + 255) & ~(size_t)255;
    return p;
  };
  const size_t LDD = (size_t)L_LAYERS * D_MODEL * D_MODEL;
  const size_t XN  = (size_t)BT * D_MODEL;

  bf16_t* Wbf   = (bf16_t*)alloc(3 * LDD * sizeof(bf16_t));  // 24 MB
  float*  x     = (float*) alloc(XN * sizeof(float));        // 64 MB
  float*  Qb    = (float*) alloc(XN * sizeof(float));        // 64 MB (also 'inter')
  float*  Rb    = (float*) alloc(XN * sizeof(float));        // 64 MB (also 'y')
  float*  Hb    = (float*) alloc(XN * sizeof(float));        // 64 MB
  bf16_t* xbf   = (bf16_t*)alloc(XN * sizeof(bf16_t));       // 32 MB
  float*  pooled= (float*) alloc(B_ * D_MODEL * sizeof(float));

  // Weights -> bf16 (recomputed every call; deterministic).
  f2bf_kernel<<<1024, 256, 0, stream>>>(Wq, Wbf + 0 * LDD, (long)LDD);
  f2bf_kernel<<<1024, 256, 0, stream>>>(Wr, Wbf + 1 * LDD, (long)LDD);
  f2bf_kernel<<<1024, 256, 0, stream>>>(Wh, Wbf + 2 * LDD, (long)LDD);

  embed_kernel<<<4096, 256, 0, stream>>>(tokens, embw, pos, x, xbf, (long)XN);

  dim3 ggrid(BT / TM, D_MODEL / TN, 3);
  for (int l = 0; l < L_LAYERS; ++l) {
    const size_t wo = (size_t)l * D_MODEL * D_MODEL;
    gemm_qrh_wmma<<<ggrid, 256, 0, stream>>>(
        xbf,
        Wbf + 0 * LDD + wo, Wbf + 1 * LDD + wo, Wbf + 2 * LDD + wo,
        bq + l * D_MODEL, br + l * D_MODEL, bh + l * D_MODEL,
        Qb, Rb, Hb, BT, D_MODEL, D_MODEL);

    spectral_kernel<<<B_ * D_MODEL, 256, 0, stream>>>(Qb, Rb, alpha + l * QD_);
    // 'inter' now lives in Qb; Rb is free -> reuse as 'y'.
    hamln_kernel<<<BT, 256, 0, stream>>>(Qb, Hb, spw + l * D_MODEL, spb + l * D_MODEL, Rb);
    grln_kernel<<<BT, 256, 0, stream>>>(Rb, rot + l * 16, gate + l * QD_,
                                        olw + l * D_MODEL, olb + l * D_MODEL, x, xbf);
  }

  pool_kernel<<<B_ * 4, 256, 0, stream>>>(x, pooled);
  cls_kernel<<<B_, 256, 0, stream>>>(pooled, clw, clb, clsW, clsb, out);
}
